// MultiHeadAttention_37495064494383
// MI455X (gfx1250) — matrix operand order repaired and checked by reading, not compiled, not run on hardware
//
#include <hip/hip_runtime.h>

// ---------------------------------------------------------------------------
// MI455X (gfx1250) fused MHA forward.
// All matmuls via v_wmma_f32_16x16x32_bf16 (wave32 WMMA, f32 accumulate).
// ---------------------------------------------------------------------------

typedef __bf16 bf16_t;
typedef __attribute__((ext_vector_type(16))) __bf16 v16bf;
typedef __attribute__((ext_vector_type(8)))  __bf16 v8bf;
typedef __attribute__((ext_vector_type(8)))  float  v8f;

#define Bsz 8
#define Tsz 1024
#define Dsz 1024
#define NH  8
#define HD  128                     // head dim
#define HB  (NH * Bsz)              // 64 (head,batch) pairs
#define NEG_BIG   (-4294967295.0f)  // -2^32+1 (rounds to -2^32 in f32, same as ref)
#define INV_SQRTD 0.08838834764831843f

#define WMMA_BF16(a, b, c) \
  __builtin_amdgcn_wmma_f32_16x16x32_bf16(false, (a), false, (b), (short)0, (c), false, false)

// ---- fragment loaders -----------------------------------------------------
// A-matrix 16x32 bf16: lane = M row (halves pick K range), elements = K.
static __device__ inline v16bf afrag_bf16(const bf16_t* base, int ld, int lane) {
  const int row = lane & 15;
  const int kb  = (lane >> 4) << 3;
  const bf16_t* p = base + (size_t)row * ld + kb;
  v8bf lo = *(const v8bf*)p;
  v8bf hi = *(const v8bf*)(p + 16);
  return __builtin_shufflevector(lo, hi, 0,1,2,3,4,5,6,7,8,9,10,11,12,13,14,15);
}

// A fragment converted from an f32 source (x rows, or LDS probability panel).
static __device__ inline v16bf afrag_f32(const float* base, int ld, int lane, float scale) {
  const int row = lane & 15;
  const int kb  = (lane >> 4) << 3;
  const float* p = base + (size_t)row * ld + kb;
  v16bf r;
#pragma unroll
  for (int i = 0; i < 8; ++i) r[i]     = (__bf16)(p[i]      * scale);
#pragma unroll
  for (int i = 0; i < 8; ++i) r[8 + i] = (__bf16)(p[16 + i] * scale);
  return r;
}

// B-matrix 32x16 bf16 from row-major (K,N) storage: lane = N column (halves pick K range), elements = K.
static __device__ inline v16bf bfrag_rm(const bf16_t* base, int ld, int lane) {
  const int kb = (lane >> 4) << 3;
  const bf16_t* p = base + (size_t)kb * ld + (lane & 15);
  v16bf r;
#pragma unroll
  for (int i = 0; i < 8; ++i) r[i]     = p[(size_t)i * ld];
#pragma unroll
  for (int i = 0; i < 8; ++i) r[8 + i] = p[(size_t)(16 + i) * ld];
  return r;
}

// ---------------------------------------------------------------------------
// Kernel 0: weight prep — Wt[k][n] = (bf16) W[n][k], LDS-tiled transpose.
// grid (32, 32, 3), block (32, 8)
// ---------------------------------------------------------------------------
__global__ __launch_bounds__(256) void wprep_kernel(
    const float* __restrict__ Wq, const float* __restrict__ Wk,
    const float* __restrict__ Wv, bf16_t* __restrict__ Wt3)
{
  __shared__ float tile[32][33];
  const int tx = threadIdx.x, ty = threadIdx.y;
  const int k0 = blockIdx.x * 32, n0 = blockIdx.y * 32;
  const int which = blockIdx.z;
  const float* W = (which == 0) ? Wq : (which == 1) ? Wk : Wv;
  bf16_t* Wt = Wt3 + (size_t)which * Dsz * Dsz;

#pragma unroll
  for (int i = 0; i < 32; i += 8)                      // read: coalesced along k
    tile[ty + i][tx] = W[(size_t)(n0 + ty + i) * Dsz + k0 + tx];
  __syncthreads();
#pragma unroll
  for (int i = 0; i < 32; i += 8)                      // write: coalesced along n
    Wt[(size_t)(k0 + ty + i) * Dsz + n0 + tx] = (bf16_t)tile[tx][ty + i];
}

// ---------------------------------------------------------------------------
// Kernel 1: Q = x Wq^T, K = x Wk^T, V = x Wv^T  (bf16 outputs, head-major)
//   Qh: (h,b,t,dd) row-major     -> WMMA A operand for scores
//   Kt: (h,b,dd,t) (transposed)  -> WMMA B operand for scores
//   Vh: (h,b,t,dd) row-major     -> WMMA B operand for P*V
// Each wave: 16x64 output strip (4 WMMA accum chains share one A fragment).
// grid (128, 16, 3), block 128 (4 waves stacked along M)
// ---------------------------------------------------------------------------
__global__ __launch_bounds__(128) void qkv_proj_kernel(
    const float* __restrict__ x, const bf16_t* __restrict__ Wt3,
    bf16_t* __restrict__ Qh, bf16_t* __restrict__ Kt, bf16_t* __restrict__ Vh)
{
  const int lane = threadIdx.x & 31;
  const int wave = threadIdx.x >> 5;
  const int m0 = blockIdx.x * 64 + wave * 16;     // token-row tile (B*T rows)
  const int n0 = blockIdx.y * 64;                 // output-feature strip
  const int which = blockIdx.z;
  const bf16_t* Wt = Wt3 + (size_t)which * Dsz * Dsz;

  v8f acc[4] = {};
  for (int k0 = 0; k0 < Dsz; k0 += 32) {
    v16bf a = afrag_f32(x + (size_t)m0 * Dsz + k0, Dsz, lane, 1.0f);
#pragma unroll
    for (int j = 0; j < 4; ++j) {
      v16bf b = bfrag_rm(Wt + (size_t)k0 * Dsz + n0 + j * 16, Dsz, lane);
      acc[j] = WMMA_BF16(a, b, acc[j]);
    }
  }

#pragma unroll
  for (int j = 0; j < 4; ++j) {
    const int n  = n0 + j * 16 + (lane & 15);
    const int h  = n >> 7;
    const int dd = n & 127;
#pragma unroll
    for (int r = 0; r < 8; ++r) {
      const int m  = m0 + r + ((lane >> 4) << 3);
      const int bi = m >> 10;
      const int t  = m & 1023;
      const int hb = h * Bsz + bi;
      const bf16_t v = (bf16_t)acc[j][r];
      if (which == 0)      Qh[((size_t)hb * Tsz + t) * HD + dd] = v;
      else if (which == 1) Kt[((size_t)hb * HD + dd) * Tsz + t] = v;
      else                 Vh[((size_t)hb * Tsz + t) * HD + dd] = v;
    }
  }
}

// ---------------------------------------------------------------------------
// Kernel 2: query/key padding masks: sign(sum |row|)
// ---------------------------------------------------------------------------
__global__ void masks_kernel(const bf16_t* __restrict__ Qh, const bf16_t* __restrict__ Kt,
                             float* __restrict__ qm, float* __restrict__ km)
{
  const int idx = blockIdx.x * blockDim.x + threadIdx.x;
  if (idx >= HB * Tsz) return;
  const int hb = idx >> 10;
  const int t  = idx & 1023;
  float sq = 0.f, sk = 0.f;
  const bf16_t* qr = Qh + (size_t)idx * HD;
  const bf16_t* kc = Kt + (size_t)hb * HD * Tsz + t;
#pragma unroll 4
  for (int i = 0; i < HD; ++i) sq += fabsf((float)qr[i]);
#pragma unroll 4
  for (int i = 0; i < HD; ++i) sk += fabsf((float)kc[(size_t)i * Tsz]);
  qm[idx] = (sq > 0.f) ? 1.f : 0.f;
  km[idx] = (sk > 0.f) ? 1.f : 0.f;
}

// ---------------------------------------------------------------------------
// Kernel 3: attention for one (h, b, 16-query tile).
// 4 waves. Full 16x1024 score panel in LDS (64KB) -> exact softmax, then P*V.
// grid (64, 8, 8), block 128
// ---------------------------------------------------------------------------
__global__ __launch_bounds__(128) void attn_kernel(
    const bf16_t* __restrict__ Qh, const bf16_t* __restrict__ Kt,
    const bf16_t* __restrict__ Vh, const float* __restrict__ qm,
    const float* __restrict__ km,  const float* __restrict__ mask,
    float* __restrict__ out)
{
  __shared__ float sc[16 * Tsz];   // 64 KB score / probability panel

  const int lane = threadIdx.x & 31;
  const int wave = threadIdx.x >> 5;
  const int q0 = blockIdx.x * 16;
  const int bi = blockIdx.y;
  const int h  = blockIdx.z;
  const int hb = h * Bsz + bi;

  // Preload Q fragments for this query tile (16 x 128).
  v16bf qa[4];
#pragma unroll
  for (int kk = 0; kk < 4; ++kk)
    qa[kk] = afrag_bf16(Qh + ((size_t)hb * Tsz + q0) * HD + kk * 32, HD, lane);

  // ---- scores = Q K^T / sqrt(d), masked --------------------------------
  for (int kt = wave; kt < Tsz / 16; kt += 4) {
    const int key0 = kt * 16;
    v8f c = {};
#pragma unroll
    for (int kk = 0; kk < 4; ++kk) {
      v16bf bk = bfrag_rm(Kt + ((size_t)hb * HD + kk * 32) * Tsz + key0, Tsz, lane);
      c = WMMA_BF16(qa[kk], bk, c);
    }
    const int key = key0 + (lane & 15);
    const float kmv = km[hb * Tsz + key];
#pragma unroll
    for (int r = 0; r < 8; ++r) {
      const int m = r + ((lane >> 4) << 3);
      float s = c[r] * INV_SQRTD;
      if (kmv == 0.f) s = NEG_BIG;
      s += (key > (q0 + m)) ? 0.f : -10000.f;   // FORWARD adder
      sc[m * Tsz + key] = s;
    }
  }
  __syncthreads();

  // ---- softmax per query row (wave handles 4 rows) ---------------------
  for (int rr = 0; rr < 4; ++rr) {
    const int m = wave * 4 + rr;
    float mx = -3.4e38f;
    for (int c = lane; c < Tsz; c += 32) mx = fmaxf(mx, sc[m * Tsz + c]);
#pragma unroll
    for (int off = 16; off > 0; off >>= 1) mx = fmaxf(mx, __shfl_xor(mx, off, 32));
    float sum = 0.f;
    for (int c = lane; c < Tsz; c += 32) sum += __expf(sc[m * Tsz + c] - mx);
#pragma unroll
    for (int off = 16; off > 0; off >>= 1) sum += __shfl_xor(sum, off, 32);
    const float scl = qm[hb * Tsz + q0 + m] / sum;   // fold query-mask into scale
    for (int c = lane; c < Tsz; c += 32)
      sc[m * Tsz + c] = __expf(sc[m * Tsz + c] - mx) * scl;
  }
  __syncthreads();

  // ---- out = P V  (wave owns 32 output columns) ------------------------
  v8f o0 = {}, o1 = {};
  const int d0 = wave * 32;
  for (int k0 = 0; k0 < Tsz; k0 += 32) {
    v16bf pa = afrag_f32(sc + k0, Tsz, lane, 1.0f);   // P fragment from LDS
    v16bf b0 = bfrag_rm(Vh + ((size_t)hb * Tsz + k0) * HD + d0,      HD, lane);
    v16bf b1 = bfrag_rm(Vh + ((size_t)hb * Tsz + k0) * HD + d0 + 16, HD, lane);
    o0 = WMMA_BF16(pa, b0, o0);
    o1 = WMMA_BF16(pa, b1, o1);
  }

  const int n = lane & 15;
#pragma unroll
  for (int r = 0; r < 8; ++r) {
    const int m  = r + ((lane >> 4) << 3);
    const int qg = q0 + m;
    const float mv = mask[bi * Tsz + qg];
    float* op = out + ((size_t)(bi * Tsz + qg)) * Dsz + h * HD + d0;
    op[n]      = o0[r] * mv;
    op[16 + n] = o1[r] * mv;
  }
}

// ---------------------------------------------------------------------------
extern "C" void kernel_launch(void* const* d_in, const int* in_sizes, int n_in,
                              void* d_out, int out_size, void* d_ws, size_t ws_size,
                              hipStream_t stream) {
  (void)in_sizes; (void)n_in; (void)out_size; (void)ws_size;
  const float* x    = (const float*)d_in[0];
  const float* mask = (const float*)d_in[1];
  const float* Wq   = (const float*)d_in[2];
  const float* Wk   = (const float*)d_in[3];
  const float* Wv   = (const float*)d_in[4];
  float* out = (float*)d_out;

  // Workspace layout (needs ~54.6 MB): bf16 Qh, Kt, Vh + f32 masks + bf16 Wt.
  const size_t HEAD_ELEMS = (size_t)HB * Tsz * HD;   // 8M elements
  bf16_t* Qh  = (bf16_t*)d_ws;
  bf16_t* Kt  = Qh + HEAD_ELEMS;
  bf16_t* Vh  = Kt + HEAD_ELEMS;
  float*  qm  = (float*)(Vh + HEAD_ELEMS);
  float*  km  = qm + (size_t)HB * Tsz;
  bf16_t* Wt3 = (bf16_t*)(km + (size_t)HB * Tsz);

  wprep_kernel<<<dim3(Dsz / 32, Dsz / 32, 3), dim3(32, 8), 0, stream>>>(
      Wq, Wk, Wv, Wt3);
  qkv_proj_kernel<<<dim3((Bsz * Tsz) / 64, Dsz / 64, 3), 128, 0, stream>>>(
      x, Wt3, Qh, Kt, Vh);
  masks_kernel<<<(HB * Tsz + 255) / 256, 256, 0, stream>>>(Qh, Kt, qm, km);
  attn_kernel<<<dim3(Tsz / 16, Bsz, NH), 128, 0, stream>>>(
      Qh, Kt, Vh, qm, km, mask, out);
}
